// LinformerSelfAttention_69449621176582
// MI455X (gfx1250) — matrix operand order repaired and checked by reading, not compiled, not run on hardware
//
#include <hip/hip_runtime.h>
#include <hip/hip_bf16.h>
#include <cmath>

typedef __bf16 bf16;
typedef __attribute__((ext_vector_type(16))) __bf16 v16bf;
typedef __attribute__((ext_vector_type(8)))  float  v8f;
typedef __attribute__((ext_vector_type(4)))  unsigned int u32x4;
typedef __attribute__((ext_vector_type(4)))  int i32x4;
typedef __attribute__((ext_vector_type(8)))  int i32x8;

#define B_  4
#define T_  4096
#define C_  1024
#define H_  16
#define D_  64
#define P_  256
#define BH_ (B_*H_)
#define BT_ (B_*T_)

#if defined(__has_builtin)
#if __has_builtin(__builtin_amdgcn_tensor_load_to_lds) && __has_builtin(__builtin_amdgcn_s_wait_tensorcnt)
#define HAVE_TDM 1
#endif
#endif

// ---------------------------------------------------------------- WMMA helpers

static __device__ __forceinline__ v8f wmma_bf16(v16bf a, v16bf b, v8f c) {
  return __builtin_amdgcn_wmma_f32_16x16x32_bf16(false, a, false, b, (short)0, c, false, false);
}

union FragU { uint4 q[2]; v16bf v; };

// A fragment (16x32 row-major): lane m = lane&15, 8 bf16 at k0+hi*8 and 8 at +16.
static __device__ __forceinline__ v16bf load_a_frag(const bf16* base, int ld, int k0, int lane) {
  const int m  = lane & 15;
  const int kb = k0 + ((lane >> 4) << 3);
  const char* r = reinterpret_cast<const char*>(base + (size_t)m * ld + kb);
  FragU u;
  u.q[0] = *reinterpret_cast<const uint4*>(r);
  u.q[1] = *reinterpret_cast<const uint4*>(r + 32);
  return u.v;
}

// B fragment (32x16) from K-contiguous [N][K]: lane n = lane&15, 8 bf16 at k0+hi*8 and 8 at +16.
static __device__ __forceinline__ v16bf load_b_frag(const bf16* base, int ld, int k0, int lane) {
  const int n  = lane & 15;
  const int kb = k0 + ((lane >> 4) << 3);
  const char* r = reinterpret_cast<const char*>(base + (size_t)n * ld + kb);
  FragU u;
  u.q[0] = *reinterpret_cast<const uint4*>(r);
  u.q[1] = *reinterpret_cast<const uint4*>(r + 32);
  return u.v;
}

static __device__ __forceinline__ uint4 pack8(v8f a) {
  union { bf16 h[8]; uint4 q; } u;
#pragma unroll
  for (int i = 0; i < 8; ++i) u.h[i] = (bf16)a[i];
  return u.q;
}

// ---------------------------------------------------------------- TDM (async tensor) helper
// 2D bf16 tile: tile == tensor block, dims in elements, stride0 in elements.
#ifdef HAVE_TDM
static __device__ __forceinline__ void tdm_load_2d_bf16(unsigned lds_off, const void* gptr,
                                                        unsigned dim0, unsigned dim1,
                                                        unsigned stride0) {
  unsigned long long ga = (unsigned long long)(size_t)gptr;
  u32x4 g0;
  g0[0] = 1u;                                            // count=1 (valid, user mode)
  g0[1] = lds_off;                                       // lds_addr [63:32]
  g0[2] = (unsigned)ga;                                  // global_addr [95:64]
  g0[3] = (unsigned)((ga >> 32) & 0x01ffffffu) | (2u << 30);  // addr[120:96], type=2
  i32x8 g1;
  g1[0] = (int)(1u << 16);                               // data_size=1 (2 bytes), mask=0
  g1[1] = (int)((dim0 & 0xffffu) << 16);                 // tensor_dim0[15:0] @ [63:48]
  g1[2] = (int)((dim0 >> 16) | ((dim1 & 0xffffu) << 16));// dim0[31:16], dim1[15:0]
  g1[3] = (int)((dim1 >> 16) | ((dim0 & 0xffffu) << 16));// dim1[31:16], tile_dim0 @ [127:112]
  g1[4] = (int)(dim1 & 0xffffu);                         // tile_dim1 @ [143:128], tile_dim2=0
  g1[5] = (int)stride0;                                  // tensor_dim0_stride[31:0]
  g1[6] = 0;                                             // stride[47:32]=0, dim1_stride lo=0
  g1[7] = 0;
  i32x4 z4 = {0, 0, 0, 0};
#if __clang_major__ >= 23
  i32x8 z8 = {0, 0, 0, 0, 0, 0, 0, 0};
  __builtin_amdgcn_tensor_load_to_lds(g0, g1, z4, z4, z8, 0);
#else
  __builtin_amdgcn_tensor_load_to_lds(g0, g1, z4, z4, 0);
#endif
}
#endif

// ---------------------------------------------------------------- prep kernels

__global__ void k_cvt_bf16(const float* __restrict__ in, bf16* __restrict__ out, int n4) {
  int i = blockIdx.x * blockDim.x + threadIdx.x;
  if (i >= n4) return;
  float4 f = reinterpret_cast<const float4*>(in)[i];
  union { bf16 h[4]; uint2 u; } p;
  p.h[0] = (bf16)f.x; p.h[1] = (bf16)f.y; p.h[2] = (bf16)f.z; p.h[3] = (bf16)f.w;
  reinterpret_cast<uint2*>(out)[i] = p.u;
}

// E [rows=T, cols=P] f32 -> E^T [P, T] bf16
__global__ void k_transpose_bf16(const float* __restrict__ E, bf16* __restrict__ ET,
                                 int rows, int cols) {
  int i = blockIdx.x * blockDim.x + threadIdx.x;
  if (i >= rows * cols) return;
  int p = i / rows, t = i % rows;
  ET[(size_t)p * rows + t] = (bf16)E[(size_t)t * cols + p];
}

// ---------------------------------------------------------------- Q projection
// C[m=feature, n=token] = sum_k Wq[m,k]*x[n,k]; 2x4 tiles/wave; store bf16 q[B,H,T,D]
__global__ __launch_bounds__(128)
void k_gemm_q(const bf16* __restrict__ Wb, const bf16* __restrict__ xb,
              bf16* __restrict__ qb) {
  const int lane = threadIdx.x & 31;
  const int wid  = blockIdx.x * (blockDim.x >> 5) + (threadIdx.x >> 5);
  const int mt2 = wid >> 8;   // 32 groups of 2 feature tiles
  const int ng  = wid & 255;  // 256 groups of 64 tokens
  const bf16* A0 = Wb + (size_t)(mt2 * 32) * C_;
  v8f acc[2][4] = {};
  for (int k0 = 0; k0 < C_; k0 += 32) {
    v16bf a0 = load_a_frag(A0, C_, k0, lane);
    v16bf a1 = load_a_frag(A0 + (size_t)16 * C_, C_, k0, lane);
#pragma unroll
    for (int j = 0; j < 4; ++j) {
      v16bf b = load_b_frag(xb + (size_t)(ng * 64 + j * 16) * C_, C_, k0, lane);
      acc[0][j] = wmma_bf16(a0, b, acc[0][j]);
      acc[1][j] = wmma_bf16(a1, b, acc[1][j]);
    }
  }
  const int hi = lane >> 4;
#pragma unroll
  for (int i = 0; i < 2; ++i) {
    const int m0 = mt2 * 32 + i * 16 + hi * 8;   // 8 consecutive features
    const int h = m0 >> 6, d0 = m0 & 63;
#pragma unroll
    for (int j = 0; j < 4; ++j) {
      const int tok = ng * 64 + j * 16 + (lane & 15);
      const int b = tok >> 12, t = tok & (T_ - 1);
      bf16* dst = qb + (((size_t)(b * H_ + h) * T_ + t) * D_ + d0);
      *reinterpret_cast<uint4*>(dst) = pack8(acc[i][j]);
    }
  }
}

// ---------------------------------------------------------------- K+V projection (fused)
// C[m=token, n=feature]; store transposed bf16 kT/vT [B,H,D,T]
__global__ __launch_bounds__(128)
void k_gemm_kv(const bf16* __restrict__ xb, const bf16* __restrict__ Wk,
               const bf16* __restrict__ Wv, bf16* __restrict__ kT,
               bf16* __restrict__ vT) {
  const int lane = threadIdx.x & 31;
  const int wid  = blockIdx.x * (blockDim.x >> 5) + (threadIdx.x >> 5);
  const int mt = wid >> 4;   // 1024 token tiles
  const int ng = wid & 15;   // 16 groups of 64 features
  const bf16* Arow = xb + (size_t)(mt * 16) * C_;
  v8f ak[4] = {}, av[4] = {};
  for (int k0 = 0; k0 < C_; k0 += 32) {
    v16bf a = load_a_frag(Arow, C_, k0, lane);
#pragma unroll
    for (int j = 0; j < 4; ++j) {
      const size_t ro = (size_t)(ng * 64 + j * 16) * C_;
      ak[j] = wmma_bf16(a, load_b_frag(Wk + ro, C_, k0, lane), ak[j]);
      av[j] = wmma_bf16(a, load_b_frag(Wv + ro, C_, k0, lane), av[j]);
    }
  }
  const int hi = lane >> 4;
  const int tok0 = mt * 16 + hi * 8;
  const int b = tok0 >> 12, t = tok0 & (T_ - 1);
#pragma unroll
  for (int j = 0; j < 4; ++j) {
    const int f = ng * 64 + j * 16 + (lane & 15);
    const int h = f >> 6, d = f & 63;
    const size_t o = ((size_t)(b * H_ + h) * D_ + d) * T_ + t;
    *reinterpret_cast<uint4*>(kT + o) = pack8(ak[j]);
    *reinterpret_cast<uint4*>(vT + o) = pack8(av[j]);
  }
}

// ---------------------------------------------------------------- low-rank projections
// per bh: C[m,n] = sum_t A[m,t]*B[n,t], K=T; 2x2 tiles/wave; out[bh][n*M+m]
__global__ __launch_bounds__(128)
void k_proj(const bf16* __restrict__ A, long aStride,
            const bf16* __restrict__ Bm, long bStride,
            bf16* __restrict__ out, long oStride, int M, int N) {
  const int lane = threadIdx.x & 31;
  const int wid  = blockIdx.x * (blockDim.x >> 5) + (threadIdx.x >> 5);
  const int ntc2 = N >> 5;
  const int tilesPer = (M >> 5) * ntc2;
  const int bh = wid / tilesPer;
  const int r  = wid % tilesPer;
  const int mt = r / ntc2, nt = r % ntc2;
  const bf16* Ab = A  + (size_t)bh * aStride + (size_t)(mt * 32) * T_;
  const bf16* Bb = Bm + (size_t)bh * bStride + (size_t)(nt * 32) * T_;
  v8f acc[2][2] = {};
  for (int k0 = 0; k0 < T_; k0 += 32) {
    v16bf a0 = load_a_frag(Ab, T_, k0, lane);
    v16bf a1 = load_a_frag(Ab + (size_t)16 * T_, T_, k0, lane);
    v16bf b0 = load_b_frag(Bb, T_, k0, lane);
    v16bf b1 = load_b_frag(Bb + (size_t)16 * T_, T_, k0, lane);
    acc[0][0] = wmma_bf16(a0, b0, acc[0][0]);
    acc[0][1] = wmma_bf16(a0, b1, acc[0][1]);
    acc[1][0] = wmma_bf16(a1, b0, acc[1][0]);
    acc[1][1] = wmma_bf16(a1, b1, acc[1][1]);
  }
#pragma unroll
  for (int i = 0; i < 2; ++i) {
#pragma unroll
    for (int j = 0; j < 2; ++j) {
      const int n  = nt * 32 + j * 16 + (lane & 15);
      const int m0 = mt * 32 + i * 16 + (lane >> 4) * 8;
      bf16* dst = out + (size_t)bh * oStride + (size_t)n * M + m0;
      *reinterpret_cast<uint4*>(dst) = pack8(acc[i][j]);
    }
  }
}

// ---------------------------------------------------------------- fused attention
// 2 waves/block share one (b,h); k_proj block TDM-staged into LDS; softmax in registers
__global__ __launch_bounds__(64)
void k_attn(const bf16* __restrict__ qb, const bf16* __restrict__ kp,
            const bf16* __restrict__ vpT, bf16* __restrict__ yb) {
  __shared__ __align__(16) bf16 KPL[P_ * D_];      // 32KB: k_proj[bh] [P][D]
  __shared__ __align__(16) bf16 ATT[2][16 * 256];  // 16KB: normalized att tiles

  const int lane = threadIdx.x & 31;
  const int w    = threadIdx.x >> 5;
  const int wid  = blockIdx.x * 2 + w;
  const int tt = wid & 255;
  const int bh = wid >> 8;
  const int t0 = tt * 16;
  const int hi = lane >> 4, lo = lane & 15;

  // stage k_proj[bh] (P_ x D_ bf16, 32KB) into LDS with the Tensor Data Mover
#ifdef HAVE_TDM
  if (w == 0) {
    tdm_load_2d_bf16((unsigned)(size_t)&KPL[0], kp + (size_t)bh * P_ * D_,
                     (unsigned)D_, (unsigned)P_, (unsigned)D_);
    __builtin_amdgcn_s_wait_tensorcnt(0);
  }
#else
  {
    const uint4* src = reinterpret_cast<const uint4*>(kp + (size_t)bh * P_ * D_);
    uint4* dst = reinterpret_cast<uint4*>(&KPL[0]);
    for (int i = threadIdx.x; i < P_ * D_ / 8; i += 64) dst[i] = src[i];
  }
#endif
  __syncthreads();

  // q A-fragments (K = D = 64 -> two 16x32 frags)
  const bf16* qrow = qb + ((size_t)bh * T_ + t0) * D_;
  const v16bf a0 = load_a_frag(qrow, D_, 0, lane);
  const v16bf a1 = load_a_frag(qrow, D_, 32, lane);

  // scores (16 n-tiles kept in registers), bucket-causal mask: keep t >= 16p+15
  v8f sc[16];
#pragma unroll
  for (int pt = 0; pt < 16; ++pt) {
    const bf16* brow = &KPL[(size_t)(pt * 16) * D_];
    v8f acc = {};
    acc = wmma_bf16(a0, load_b_frag(brow, D_, 0, lane), acc);
    acc = wmma_bf16(a1, load_b_frag(brow, D_, 32, lane), acc);
    const int p = pt * 16 + lo;
#pragma unroll
    for (int g = 0; g < 8; ++g) {
      const int t = t0 + g + hi * 8;
      acc[g] = (t < 16 * p + 15) ? -INFINITY : acc[g] * 0.125f;
    }
    sc[pt] = acc;
  }

  // register softmax: rows r = g + 8*hi; reduce over p across the 16-lane half
  float mx[8], sm[8];
#pragma unroll
  for (int g = 0; g < 8; ++g) mx[g] = -INFINITY;
#pragma unroll
  for (int pt = 0; pt < 16; ++pt)
#pragma unroll
    for (int g = 0; g < 8; ++g) mx[g] = fmaxf(mx[g], sc[pt][g]);
#pragma unroll
  for (int msk = 1; msk < 16; msk <<= 1)
#pragma unroll
    for (int g = 0; g < 8; ++g) mx[g] = fmaxf(mx[g], __shfl_xor(mx[g], msk, 32));
#pragma unroll
  for (int g = 0; g < 8; ++g) sm[g] = 0.f;
#pragma unroll
  for (int pt = 0; pt < 16; ++pt)
#pragma unroll
    for (int g = 0; g < 8; ++g) {
      float e = (mx[g] == -INFINITY) ? 0.f : __expf(sc[pt][g] - mx[g]);
      sc[pt][g] = e;
      sm[g] += e;
    }
#pragma unroll
  for (int msk = 1; msk < 16; msk <<= 1)
#pragma unroll
    for (int g = 0; g < 8; ++g) sm[g] += __shfl_xor(sm[g], msk, 32);
#pragma unroll
  for (int g = 0; g < 8; ++g) mx[g] = (sm[g] > 0.f) ? 1.f / sm[g] : 0.f;  // reuse as rinv

  // normalized bf16 att tile -> LDS (A-operand relayout for the AV product)
#pragma unroll
  for (int pt = 0; pt < 16; ++pt)
#pragma unroll
    for (int g = 0; g < 8; ++g)
      ATT[w][(g + hi * 8) * 256 + pt * 16 + lo] = (bf16)(sc[pt][g] * mx[g]);
  __syncthreads();

  // y[t,d] = sum_p att[t,p] * v_projT[d,p]   (K = P = 256)
  const bf16* vpb = vpT + (size_t)bh * D_ * P_;
  const int h = bh & (H_ - 1), b = bh >> 4;
#pragma unroll
  for (int dt = 0; dt < 4; ++dt) {
    v8f acc = {};
#pragma unroll
    for (int ks = 0; ks < 256; ks += 32) {
      v16bf aF = load_a_frag(&ATT[w][0], 256, ks, lane);
      v16bf bF = load_b_frag(vpb + (size_t)(dt * 16) * P_, P_, ks, lane);
      acc = wmma_bf16(aF, bF, acc);
    }
    const int col = h * D_ + dt * 16 + lo;
#pragma unroll
    for (int g = 0; g < 8; ++g)
      yb[(size_t)(b * T_ + t0 + g + hi * 8) * C_ + col] = (bf16)acc[g];
  }
}

// ---------------------------------------------------------------- output projection
// C[m=out-feature, n=token] = sum_k Wo[m,k]*y[n,k]; 2x4 tiles/wave; f32 out [BT,C]
__global__ __launch_bounds__(128)
void k_gemm_out(const bf16* __restrict__ Wob, const bf16* __restrict__ yb,
                float* __restrict__ out) {
  const int lane = threadIdx.x & 31;
  const int wid  = blockIdx.x * (blockDim.x >> 5) + (threadIdx.x >> 5);
  const int mt2 = wid >> 8;
  const int ng  = wid & 255;
  const bf16* A0 = Wob + (size_t)(mt2 * 32) * C_;
  v8f acc[2][4] = {};
  for (int k0 = 0; k0 < C_; k0 += 32) {
    v16bf a0 = load_a_frag(A0, C_, k0, lane);
    v16bf a1 = load_a_frag(A0 + (size_t)16 * C_, C_, k0, lane);
#pragma unroll
    for (int j = 0; j < 4; ++j) {
      v16bf b = load_b_frag(yb + (size_t)(ng * 64 + j * 16) * C_, C_, k0, lane);
      acc[0][j] = wmma_bf16(a0, b, acc[0][j]);
      acc[1][j] = wmma_bf16(a1, b, acc[1][j]);
    }
  }
  const int hi = lane >> 4;
#pragma unroll
  for (int i = 0; i < 2; ++i) {
    const int c0 = mt2 * 32 + i * 16 + hi * 8;
#pragma unroll
    for (int j = 0; j < 4; ++j) {
      const int tok = ng * 64 + j * 16 + (lane & 15);
      float* dst = out + (size_t)tok * C_ + c0;
      reinterpret_cast<float4*>(dst)[0] = make_float4(acc[i][j][0], acc[i][j][1], acc[i][j][2], acc[i][j][3]);
      reinterpret_cast<float4*>(dst)[1] = make_float4(acc[i][j][4], acc[i][j][5], acc[i][j][6], acc[i][j][7]);
    }
  }
}

// ---------------------------------------------------------------- launch

extern "C" void kernel_launch(void* const* d_in, const int* in_sizes, int n_in,
                              void* d_out, int out_size, void* d_ws, size_t ws_size,
                              hipStream_t stream) {
  (void)in_sizes; (void)n_in; (void)out_size; (void)ws_size;
  const float* x  = (const float*)d_in[0];
  const float* Wq = (const float*)d_in[1];
  const float* Wk = (const float*)d_in[2];
  const float* Wv = (const float*)d_in[3];
  const float* Wo = (const float*)d_in[4];
  const float* Ek = (const float*)d_in[5];
  const float* Ev = (const float*)d_in[6];
  float* out = (float*)d_out;

  char* ws = (char*)d_ws;
  size_t off = 0;
  auto carve = [&](size_t bytes) {
    char* p = ws + off;
    off += (bytes + 255) & ~(size_t)255;
    return p;
  };
  bf16* xb  = (bf16*)carve((size_t)BT_ * C_ * 2);
  bf16* Wqb = (bf16*)carve((size_t)C_ * C_ * 2);
  bf16* Wkb = (bf16*)carve((size_t)C_ * C_ * 2);
  bf16* Wvb = (bf16*)carve((size_t)C_ * C_ * 2);
  bf16* Wob = (bf16*)carve((size_t)C_ * C_ * 2);
  bf16* EkT = (bf16*)carve((size_t)P_ * T_ * 2);
  bf16* EvT = (bf16*)carve((size_t)P_ * T_ * 2);
  bf16* qb  = (bf16*)carve((size_t)BH_ * T_ * D_ * 2);  // [B,H,T,D]
  bf16* kT  = (bf16*)carve((size_t)BH_ * D_ * T_ * 2);  // [B,H,D,T]
  bf16* vT  = (bf16*)carve((size_t)BH_ * D_ * T_ * 2);  // [B,H,D,T]
  bf16* kp  = (bf16*)carve((size_t)BH_ * P_ * D_ * 2);  // [B,H,P,D]
  bf16* vpT = (bf16*)carve((size_t)BH_ * D_ * P_ * 2);  // [B,H,D,P]
  bf16* yb  = (bf16*)carve((size_t)BT_ * C_ * 2);       // [BT,C]

  // fp32 -> bf16
  k_cvt_bf16<<<(BT_ * C_ / 4) / 256, 256, 0, stream>>>(x, xb, BT_ * C_ / 4);
  k_cvt_bf16<<<(C_ * C_ / 4) / 256, 256, 0, stream>>>(Wq, Wqb, C_ * C_ / 4);
  k_cvt_bf16<<<(C_ * C_ / 4) / 256, 256, 0, stream>>>(Wk, Wkb, C_ * C_ / 4);
  k_cvt_bf16<<<(C_ * C_ / 4) / 256, 256, 0, stream>>>(Wv, Wvb, C_ * C_ / 4);
  k_cvt_bf16<<<(C_ * C_ / 4) / 256, 256, 0, stream>>>(Wo, Wob, C_ * C_ / 4);
  k_transpose_bf16<<<(P_ * T_) / 256, 256, 0, stream>>>(Ek, EkT, T_, P_);
  k_transpose_bf16<<<(P_ * T_) / 256, 256, 0, stream>>>(Ev, EvT, T_, P_);

  // projections
  k_gemm_q <<<2048, 128, 0, stream>>>(Wqb, xb, qb);
  k_gemm_kv<<<4096, 128, 0, stream>>>(xb, Wkb, Wvb, kT, vT);

  // k_proj[bh][p*D+d] = sum_t kT[bh][d,t] * EkT[p,t]     (M=D, N=P)
  k_proj<<<256, 128, 0, stream>>>(kT, (long)D_ * T_, EkT, 0L,
                                  kp, (long)P_ * D_, D_, P_);
  // v_projT[bh][d*P+p] = sum_t EvT[p,t] * vT[bh][d,t]    (M=P, N=D)
  k_proj<<<256, 128, 0, stream>>>(EvT, 0L, vT, (long)D_ * T_,
                                  vpT, (long)D_ * P_, P_, D_);

  // fused scores + softmax + AV
  k_attn<<<BH_ * (T_ / 16) / 2, 64, 0, stream>>>(qb, kp, vpT, yb);

  // output projection (f32)
  k_gemm_out<<<2048, 128, 0, stream>>>(Wob, yb, out);
}
